// FlaxMistralDecoderLayer_32246614459238
// MI455X (gfx1250) — compile-verified, matrix operand order checked by reading, not run on hardware
//
#include <hip/hip_runtime.h>
#include <hip/hip_bf16.h>

// ---------------------------------------------------------------------------
// Types for CDNA5 WMMA (gfx1250, wave32)
// ---------------------------------------------------------------------------
typedef __attribute__((ext_vector_type(16))) __bf16 v16bf;
typedef __attribute__((ext_vector_type(8)))  float  v8f;

__device__ __forceinline__ v8f vzero8() {
    v8f z = {0.f, 0.f, 0.f, 0.f, 0.f, 0.f, 0.f, 0.f};
    return z;
}

__device__ __forceinline__ v8f wmma_bf16(v16bf a, v16bf b, v8f c) {
    // D = A(16x32 bf16) * B(32x16 bf16) + C(16x16 f32)
    return __builtin_amdgcn_wmma_f32_16x16x32_bf16(
        /*neg_a=*/false, a, /*neg_b=*/false, b,
        /*c_mod=*/(short)0, c, /*reuse_a=*/false, /*reuse_b=*/false);
}

__device__ __forceinline__ float rmax16(float v) {
#pragma unroll
    for (int m = 1; m < 16; m <<= 1) v = fmaxf(v, __shfl_xor(v, m, 32));
    return v;
}
__device__ __forceinline__ float rsum16(float v) {
#pragma unroll
    for (int m = 1; m < 16; m <<= 1) v += __shfl_xor(v, m, 32);
    return v;
}

// ---------------------------------------------------------------------------
// Model constants
// ---------------------------------------------------------------------------
#define SEQ    2048
#define DMODEL 4096
#define FFN    14336
#define NHEADS 32
#define NKV    8
#define HDIM   128
#define KVW    (NKV * HDIM)   // 1024

// ---------------------------------------------------------------------------
// RMSNorm: fp32 (row) -> bf16, one block per row
// ---------------------------------------------------------------------------
__global__ __launch_bounds__(256) void rmsnorm_kernel(
    const float* __restrict__ X, const float* __restrict__ W,
    __bf16* __restrict__ Y, int D)
{
    const int row = blockIdx.x;
    const float* x = X + (size_t)row * D;
    float ss = 0.f;
    for (int i = threadIdx.x; i < D; i += 256) {
        float v = x[i];
        ss += v * v;
    }
    __shared__ float red[256];
    red[threadIdx.x] = ss;
    __syncthreads();
#pragma unroll
    for (int s = 128; s > 0; s >>= 1) {
        if (threadIdx.x < s) red[threadIdx.x] += red[threadIdx.x + s];
        __syncthreads();
    }
    const float inv = rsqrtf(red[0] / (float)D + 1e-5f);
    for (int i = threadIdx.x; i < D; i += 256)
        Y[(size_t)row * D + i] = (__bf16)(W[i] * x[i] * inv);
}

// ---------------------------------------------------------------------------
// RoPE: fp32 (S x H*128) -> bf16, pairwise rotation (d, d+64)
// ---------------------------------------------------------------------------
__global__ __launch_bounds__(256) void rope_kernel(
    const float* __restrict__ X, __bf16* __restrict__ Y, int H)
{
    const size_t idx = (size_t)blockIdx.x * 256 + threadIdx.x; // S*H*64 total
    const int d = (int)(idx & 63);
    const size_t t = idx >> 6;
    const int h = (int)(t % H);
    const size_t pos = t / H;
    const float a = (float)pos * __powf(10000.0f, -(float)d * (1.0f / 64.0f));
    float s, c;
    __sincosf(a, &s, &c);
    const size_t base = pos * (size_t)(H * HDIM) + (size_t)h * HDIM + d;
    const float x1 = X[base];
    const float x2 = X[base + 64];
    Y[base]      = (__bf16)(x1 * c - x2 * s);
    Y[base + 64] = (__bf16)(x2 * c + x1 * s);
}

// ---------------------------------------------------------------------------
// Tiled WMMA GEMM: C[M,N] = A(bf16, MxK row-major) @ B(fp32, KxN row-major)
// Block tile 128x128x32, 8 waves, wave tile 64x32 (4x2 WMMA 16x16 tiles).
// Double-buffered LDS; A tile staged with gfx1250 global_load_async_to_lds,
// B converted fp32->bf16 in flight (weights stream from HBM once).
// EPI: 0 = f32 store, 1 = f32 store + Res, 2 = bf16( silu(Res)*acc ),
//      3 = bf16 store.
// ---------------------------------------------------------------------------
template <int EPI>
__global__ __launch_bounds__(256) void gemm_bf16_kernel(
    const __bf16* __restrict__ A, const float* __restrict__ B,
    void* __restrict__ Cout, const float* __restrict__ Res,
    int M, int N, int K)
{
    __shared__ __align__(16) __bf16 As[2][128][40]; // [m][k], pad 8
    __shared__ __align__(16) __bf16 Bs[2][128][40]; // [n][k] transposed, pad 8

    const int tid  = threadIdx.x;
    const int lane = tid & 31;
    const int w    = tid >> 5;
    const int lo   = lane & 15;
    const int hi   = lane >> 4;
    const int wm   = w >> 2;  // 0..1
    const int wn   = w & 3;   // 0..3
    const int m0   = blockIdx.y * 128;
    const int n0   = blockIdx.x * 128;

    v8f acc[4][2];
#pragma unroll
    for (int i = 0; i < 4; ++i)
#pragma unroll
        for (int j = 0; j < 2; ++j) acc[i][j] = vzero8();

    const int arow = tid >> 1, aseg = (tid & 1) * 16;
    const int brow = tid >> 3, bseg = (tid & 7) * 16;

    auto stage = [&](int buf, int k0) {
        // --- A tile: async global->LDS (ASYNCcnt path, gfx1250) ---
        const __bf16* ap = A + (size_t)(m0 + arow) * K + k0 + aseg;
        unsigned la = (unsigned)(uintptr_t)&As[buf][arow][aseg];
        asm volatile("global_load_async_to_lds_b128 %0, %1, off"
                     :: "v"(la), "v"(ap) : "memory");
        asm volatile("global_load_async_to_lds_b128 %0, %1, off"
                     :: "v"(la + 16u), "v"(ap + 8) : "memory");
        // --- B tile: fp32 -> bf16, transposed into LDS ---
        const float* bp = B + (size_t)(k0 + brow) * N + n0 + bseg;
#pragma unroll
        for (int e = 0; e < 16; e += 4) {
            float4 f = *(const float4*)(bp + e);
            Bs[buf][bseg + e + 0][brow] = (__bf16)f.x;
            Bs[buf][bseg + e + 1][brow] = (__bf16)f.y;
            Bs[buf][bseg + e + 2][brow] = (__bf16)f.z;
            Bs[buf][bseg + e + 3][brow] = (__bf16)f.w;
        }
        __builtin_prefetch(bp + (size_t)32 * N, 0, 1);
    };

    int p = 0;
    stage(0, 0);
    asm volatile("s_wait_asynccnt 0x0" ::: "memory");
    __syncthreads();

    for (int k0 = 0; k0 < K; k0 += 32) {
        if (k0 + 32 < K) stage(1 - p, k0 + 32);

        // --- fragments + WMMA from buffer p ---
        v16bf bfr[2];
#pragma unroll
        for (int nj = 0; nj < 2; ++nj) {
            const int n = wn * 32 + nj * 16 + lo;
            ((uint4*)&bfr[nj])[0] = *(const uint4*)&Bs[p][n][hi * 8];
            ((uint4*)&bfr[nj])[1] = *(const uint4*)&Bs[p][n][hi * 8 + 16];
        }
#pragma unroll
        for (int mi = 0; mi < 4; ++mi) {
            v16bf afr;
            const int m = wm * 64 + mi * 16 + lo;
            ((uint4*)&afr)[0] = *(const uint4*)&As[p][m][hi * 8];
            ((uint4*)&afr)[1] = *(const uint4*)&As[p][m][hi * 8 + 16];
            acc[mi][0] = wmma_bf16(afr, bfr[0], acc[mi][0]);
            acc[mi][1] = wmma_bf16(afr, bfr[1], acc[mi][1]);
        }
        asm volatile("s_wait_asynccnt 0x0" ::: "memory");
        __syncthreads();
        p ^= 1;
    }

    // --- epilogue: C layout -> lane lo = col, VGPR r = row r + 8*hi ---
#pragma unroll
    for (int mi = 0; mi < 4; ++mi)
#pragma unroll
        for (int nj = 0; nj < 2; ++nj)
#pragma unroll
            for (int r = 0; r < 8; ++r) {
                const int row = m0 + wm * 64 + mi * 16 + r + 8 * hi;
                const int col = n0 + wn * 32 + nj * 16 + lo;
                const size_t idx = (size_t)row * N + col;
                const float v = acc[mi][nj][r];
                if (EPI == 0) {
                    ((float*)Cout)[idx] = v;
                } else if (EPI == 1) {
                    ((float*)Cout)[idx] = v + Res[idx];
                } else if (EPI == 2) {
                    const float g = Res[idx];
                    const float s = g / (1.f + __expf(-g));
                    ((__bf16*)Cout)[idx] = (__bf16)(s * v);
                } else {
                    ((__bf16*)Cout)[idx] = (__bf16)v;
                }
            }
}

// ---------------------------------------------------------------------------
// Causal flash attention with online softmax (GQA 32q/8kv, hd=128).
// Block = 8 waves x 32 lanes, 128 q rows; all waves share one key loop so
// the 32x128 V tile is cooperatively staged (transposed) into LDS once per
// key block. QK^T and P@V both via v_wmma_f32_16x16x32_bf16.
// ---------------------------------------------------------------------------
__global__ __launch_bounds__(256) void attn_kernel(
    const __bf16* __restrict__ Q,  // S x 4096 (head-major)
    const __bf16* __restrict__ K,  // S x 1024
    const __bf16* __restrict__ V,  // S x 1024
    __bf16* __restrict__ O)        // S x 4096
{
    const int tid  = threadIdx.x;
    const int w    = tid >> 5;
    const int lane = tid & 31;
    const int lo   = lane & 15;
    const int hi   = lane >> 4;
    const int head = blockIdx.y;
    const int kvh  = head >> 2;
    const int qb0  = blockIdx.x * 128;
    const int qw   = qb0 + w * 16;          // this wave's first q row

    __shared__ __align__(16) __bf16 Vt[128][40];   // [dim][key], pad 8
    __shared__ __align__(16) __bf16 Ps[8][16][40]; // per-wave P tile

    // preload Q fragments (16 rows x 128, 4 chunks of K=32)
    v16bf qf[4];
    {
        const __bf16* qp = Q + (size_t)(qw + lo) * DMODEL + head * HDIM;
#pragma unroll
        for (int c = 0; c < 4; ++c) {
            const int kb = c * 32 + hi * 8;
            ((uint4*)&qf[c])[0] = *(const uint4*)(qp + kb);
            ((uint4*)&qf[c])[1] = *(const uint4*)(qp + kb + 16);
        }
    }

    v8f acc[8];
#pragma unroll
    for (int i = 0; i < 8; ++i) acc[i] = vzero8();
    float rm[8], rl[8];
#pragma unroll
    for (int r = 0; r < 8; ++r) { rm[r] = -1e30f; rl[r] = 0.f; }

    const float scale = 0.08838834764831845f; // 1/sqrt(128)

    // V staging role for this thread: 8 threads per key row, 16 dims each
    const int vkey = tid >> 3;          // 0..31
    const int vseg = (tid & 7) * 16;    // 0..112

    for (int j = 0; j < qb0 + 128; j += 32) {
        // ---- cooperative stage of V tile (32 keys x 128 dims), transposed
        {
            const __bf16* vp = V + (size_t)(j + vkey) * KVW + kvh * HDIM + vseg;
            __bf16 vals[16];
            *(uint4*)&vals[0] = *(const uint4*)vp;
            *(uint4*)&vals[8] = *(const uint4*)(vp + 8);
#pragma unroll
            for (int e = 0; e < 16; ++e) Vt[vseg + e][vkey] = vals[e];
        }
        __syncthreads();

        if (j < qw + 16) {  // wave-uniform causal skip
            // ---- scores: two 16x16 tiles (keys j..j+15, j+16..j+31) ----
            v8f s0 = vzero8(), s1 = vzero8();
            {
                const __bf16* kp = K + (size_t)(j + lo) * KVW + kvh * HDIM;
                v16bf kb[4];
#pragma unroll
                for (int c = 0; c < 4; ++c) {
                    const int off = c * 32 + hi * 8;
                    ((uint4*)&kb[c])[0] = *(const uint4*)(kp + off);
                    ((uint4*)&kb[c])[1] = *(const uint4*)(kp + off + 16);
                }
#pragma unroll
                for (int c = 0; c < 4; ++c) s0 = wmma_bf16(qf[c], kb[c], s0);
                kp += (size_t)16 * KVW;
#pragma unroll
                for (int c = 0; c < 4; ++c) {
                    const int off = c * 32 + hi * 8;
                    ((uint4*)&kb[c])[0] = *(const uint4*)(kp + off);
                    ((uint4*)&kb[c])[1] = *(const uint4*)(kp + off + 16);
                }
#pragma unroll
                for (int c = 0; c < 4; ++c) s1 = wmma_bf16(qf[c], kb[c], s1);
            }

            // ---- online softmax update ----
#pragma unroll
            for (int r = 0; r < 8; ++r) {
                const int row = qw + r + 8 * hi;
                float v0 = s0[r] * scale;
                float v1 = s1[r] * scale;
                if (j + lo > row)      v0 = -1e30f;
                if (j + 16 + lo > row) v1 = -1e30f;
                float mx = rmax16(fmaxf(v0, v1));
                const float mnew  = fmaxf(rm[r], mx);
                const float alpha = __expf(rm[r] - mnew);
                const float e0 = __expf(v0 - mnew);
                const float e1 = __expf(v1 - mnew);
                rl[r] = rl[r] * alpha + rsum16(e0 + e1);
                rm[r] = mnew;
#pragma unroll
                for (int nc = 0; nc < 8; ++nc) acc[nc][r] *= alpha;
                Ps[w][r + 8 * hi][lo]      = (__bf16)e0;
                Ps[w][r + 8 * hi][lo + 16] = (__bf16)e1;
            }
            asm volatile("s_wait_dscnt 0x0" ::: "memory");

            // ---- P (16x32) as A-fragment from LDS ----
            v16bf pf;
            ((uint4*)&pf)[0] = *(const uint4*)&Ps[w][lo][hi * 8];
            ((uint4*)&pf)[1] = *(const uint4*)&Ps[w][lo][hi * 8 + 16];

            // ---- P @ V from LDS-transposed V tile ----
#pragma unroll
            for (int nc = 0; nc < 8; ++nc) {
                v16bf vf;
                ((uint4*)&vf)[0] = *(const uint4*)&Vt[nc * 16 + lo][hi * 8];
                ((uint4*)&vf)[1] = *(const uint4*)&Vt[nc * 16 + lo][hi * 8 + 16];
                acc[nc] = wmma_bf16(pf, vf, acc[nc]);
            }
        }
        __syncthreads();
    }

    // ---- normalize + store ----
#pragma unroll
    for (int nc = 0; nc < 8; ++nc)
#pragma unroll
        for (int r = 0; r < 8; ++r) {
            const int row = qw + r + 8 * hi;
            O[(size_t)row * DMODEL + head * HDIM + nc * 16 + lo] =
                (__bf16)(acc[nc][r] / rl[r]);
        }
}

// ---------------------------------------------------------------------------
// Host driver
// ---------------------------------------------------------------------------
extern "C" void kernel_launch(void* const* d_in, const int* in_sizes, int n_in,
                              void* d_out, int out_size, void* d_ws, size_t ws_size,
                              hipStream_t stream)
{
    const float* x      = (const float*)d_in[0];
    const float* wq     = (const float*)d_in[1];
    const float* wk     = (const float*)d_in[2];
    const float* wv     = (const float*)d_in[3];
    const float* wo     = (const float*)d_in[4];
    const float* ln1    = (const float*)d_in[5];
    const float* ln2    = (const float*)d_in[6];
    const float* wgate  = (const float*)d_in[7];
    const float* wup    = (const float*)d_in[8];
    const float* wdown  = (const float*)d_in[9];
    float* out = (float*)d_out;

    char* ws = (char*)d_ws;
    auto alloc = [&](size_t bytes) -> void* {
        void* p = (void*)ws;
        ws += (bytes + 255) & ~(size_t)255;
        return p;
    };
    __bf16* xn  = (__bf16*)alloc((size_t)SEQ * DMODEL * 2);   // normed x
    float*  qf  = (float*) alloc((size_t)SEQ * DMODEL * 4);   // q pre-rope
    float*  kf  = (float*) alloc((size_t)SEQ * KVW * 4);      // k pre-rope
    __bf16* qr  = (__bf16*)alloc((size_t)SEQ * DMODEL * 2);   // q roped
    __bf16* kr  = (__bf16*)alloc((size_t)SEQ * KVW * 2);      // k roped
    __bf16* vb  = (__bf16*)alloc((size_t)SEQ * KVW * 2);      // v bf16
    __bf16* ao  = (__bf16*)alloc((size_t)SEQ * DMODEL * 2);   // attn out
    float*  h   = (float*) alloc((size_t)SEQ * DMODEL * 4);   // residual 1
    __bf16* mb  = (__bf16*)alloc((size_t)SEQ * DMODEL * 2);   // normed h
    float*  gf  = (float*) alloc((size_t)SEQ * FFN * 4);      // gate
    __bf16* gu  = (__bf16*)alloc((size_t)SEQ * FFN * 2);      // silu(g)*u

    // 1) pre-attention RMSNorm
    rmsnorm_kernel<<<SEQ, 256, 0, stream>>>(x, ln1, xn, DMODEL);

    // 2) Q/K/V projections
    gemm_bf16_kernel<0><<<dim3(DMODEL / 128, SEQ / 128), 256, 0, stream>>>(
        xn, wq, qf, nullptr, SEQ, DMODEL, DMODEL);
    gemm_bf16_kernel<0><<<dim3(KVW / 128, SEQ / 128), 256, 0, stream>>>(
        xn, wk, kf, nullptr, SEQ, KVW, DMODEL);
    gemm_bf16_kernel<3><<<dim3(KVW / 128, SEQ / 128), 256, 0, stream>>>(
        xn, wv, vb, nullptr, SEQ, KVW, DMODEL);

    // 3) RoPE
    rope_kernel<<<(SEQ * NHEADS * 64) / 256, 256, 0, stream>>>(qf, qr, NHEADS);
    rope_kernel<<<(SEQ * NKV * 64) / 256, 256, 0, stream>>>(kf, kr, NKV);

    // 4) causal flash attention
    attn_kernel<<<dim3(SEQ / 128, NHEADS), 256, 0, stream>>>(qr, kr, vb, ao);

    // 5) output projection + residual
    gemm_bf16_kernel<1><<<dim3(DMODEL / 128, SEQ / 128), 256, 0, stream>>>(
        ao, wo, h, x, SEQ, DMODEL, DMODEL);

    // 6) post-attention RMSNorm
    rmsnorm_kernel<<<SEQ, 256, 0, stream>>>(h, ln2, mb, DMODEL);

    // 7) MLP: gate, up (fused silu*up), down + residual
    gemm_bf16_kernel<0><<<dim3(FFN / 128, SEQ / 128), 256, 0, stream>>>(
        mb, wgate, gf, nullptr, SEQ, FFN, DMODEL);
    gemm_bf16_kernel<2><<<dim3(FFN / 128, SEQ / 128), 256, 0, stream>>>(
        mb, wup, gu, gf, SEQ, FFN, DMODEL);
    gemm_bf16_kernel<1><<<dim3(DMODEL / 128, SEQ / 128), 256, 0, stream>>>(
        gu, wdown, out, h, SEQ, DMODEL, FFN);
}
